// PointNet2Seg_5583457485364
// MI455X (gfx1250) — matrix operand order repaired and checked by reading, not compiled, not run on hardware
//
#include <hip/hip_runtime.h>
#include <hip/hip_bf16.h>

typedef __attribute__((ext_vector_type(16))) _Float16 v16h;
typedef __attribute__((ext_vector_type(8)))  float    v8f;

#define WG 256

// ---------------------------------------------------------------------------
// Utility kernels
// ---------------------------------------------------------------------------
__global__ __launch_bounds__(WG) void wcvt_kernel(const float* __restrict__ s,
                                                  _Float16* __restrict__ d, int n) {
  int t = blockIdx.x * WG + threadIdx.x;
  if (t < n) d[t] = (_Float16)s[t];
}

__global__ __launch_bounds__(WG) void copyh_kernel(const _Float16* __restrict__ s,
                                                   _Float16* __restrict__ d, int n) {
  int t = blockIdx.x * WG + threadIdx.x;
  if (t < n) d[t] = s[t];
}

// pts (B,N,6) -> feats0 f16 (3, B*N)
__global__ __launch_bounds__(WG) void feats_split_kernel(const float* __restrict__ pts,
                                                         _Float16* __restrict__ fo,
                                                         int BN, int total) {
  int t = blockIdx.x * WG + threadIdx.x;
  if (t >= total) return;
  int c = t / BN, s = t % BN;
  fo[t] = (_Float16)pts[(long long)s * 6 + 3 + c];
}

// centers[b,i,:] = src[b, trunc(i*(Nn-1)/(M-1)), 0:3]   (src row stride = `stride`)
__global__ __launch_bounds__(WG) void centers_kernel(const float* __restrict__ src, int stride,
                                                     float* __restrict__ dst,
                                                     int M, int Nn, int total) {
  int t = blockIdx.x * WG + threadIdx.x;
  if (t >= total) return;
  int b = t / M, i = t % M;
  float step = (float)(Nn - 1) / (float)(M - 1);
  int id = (int)((float)i * step);
  if (id > Nn - 1) id = Nn - 1;
  const float* sp = src + ((long long)b * Nn + id) * stride;
  float* dp = dst + (long long)t * 3;
  dp[0] = sp[0]; dp[1] = sp[1]; dp[2] = sp[2];
}

// Brute-force KNN, one thread per query, register-resident insertion sort.
template <int K>
__global__ __launch_bounds__(WG) void knn_kernel(const float* __restrict__ q, int qs,
                                                 const float* __restrict__ p, int ps,
                                                 int* __restrict__ oi, float* __restrict__ od,
                                                 int M, int Np, int total) {
  int t = blockIdx.x * WG + threadIdx.x;
  if (t >= total) return;
  int b = t / M;
  const float* qp = q + (long long)t * qs;
  float qx = qp[0], qy = qp[1], qz = qp[2];
  const float* pb = p + (long long)b * Np * ps;
  float bd[K]; int bi[K];
#pragma unroll
  for (int j = 0; j < K; ++j) { bd[j] = 3.0e38f; bi[j] = 0; }
  for (int n = 0; n < Np; ++n) {
    const float* pp = pb + (long long)n * ps;
    float dx = qx - pp[0], dy = qy - pp[1], dz = qz - pp[2];
    float d2 = dx * dx + dy * dy + dz * dz;
    if (d2 < bd[K - 1]) {
      int pos = K - 1;
#pragma unroll
      for (int j = K - 1; j > 0; --j) {
        if (bd[j - 1] > d2) { bd[j] = bd[j - 1]; bi[j] = bi[j - 1]; pos = j - 1; }
      }
      bd[pos] = d2; bi[pos] = n;
    }
  }
#pragma unroll
  for (int j = 0; j < K; ++j) {
    oi[(long long)t * K + j] = bi[j];
    if (od) od[(long long)t * K + j] = sqrtf(fmaxf(bd[j], 1e-12f));
  }
}

// Build x_cat (Cin x Stot) f16, Cin = 3 + Cf ; Stot = B*M*32.
__global__ __launch_bounds__(WG) void sa_concat_kernel(const float* __restrict__ p, int ps,
                                                       const float* __restrict__ ctr,
                                                       const _Float16* __restrict__ feat,
                                                       const int* __restrict__ knn,
                                                       _Float16* __restrict__ xo,
                                                       int M, int Np, int BNp, int Stot,
                                                       long long total) {
  long long t = (long long)blockIdx.x * WG + threadIdx.x;
  if (t >= total) return;
  int c = (int)(t / Stot);
  int s = (int)(t % Stot);
  int k = s & 31;
  int bm = s >> 5;
  int b = bm / M;
  int n = knn[(long long)bm * 32 + k];
  float v;
  if (c < 3) {
    v = p[((long long)b * Np + n) * ps + c] - ctr[(long long)bm * 3 + c];
  } else {
    v = (float)feat[(long long)(c - 3) * BNp + (long long)b * Np + n];
  }
  xo[t] = (_Float16)v;
}

// ---------------------------------------------------------------------------
// WMMA GEMM: Y(OxS) = f16( W(OxC) * X(CxS) + bias ), f32 accumulate.
// Workgroup 256 = 8 wave32s as 2(O) x 4(S); wave tile 32x32 = 4 WMMA / K-step.
// Workgroup tile 64(O) x 128(S); K over C in steps of 32.
// LDS: A row-major (O x K), B stored TRANSPOSED (S x K) so each lane's WMMA
// fragment is contiguous -> 2x ds_load_b128 per fragment. Register double
// buffering overlaps global loads of tile t+1 with compute on tile t, plus
// global_prefetch of tile t+2.
// ---------------------------------------------------------------------------
__global__ __launch_bounds__(WG) void gemm_bias_kernel(const _Float16* __restrict__ Wt,
                                                       const _Float16* __restrict__ X,
                                                       const float* __restrict__ bias,
                                                       _Float16* __restrict__ Y,
                                                       int O, int C, int S) {
  __shared__ alignas(16) _Float16 As[2][64][40];    // [buf][o][k]   (pad 8)
  __shared__ alignas(16) _Float16 BsT[2][128][40];  // [buf][s][k]   (pad 8)

  const int tid = threadIdx.x;
  const int lane = tid & 31;
  const int wv = tid >> 5;          // 0..7
  const int owv = wv >> 2;          // 0..1  wave O tile
  const int swv = wv & 3;           // 0..3  wave S tile
  const int lm = lane & 15, gg = lane >> 4;
  const int o0 = blockIdx.y * 64;
  const int s0 = blockIdx.x * 128;

  // cooperative-load mapping
  const int arow = tid >> 2;            // 0..63
  const int acol = (tid & 3) * 8;       // 0,8,16,24
  const int cblk = tid >> 5;            // 0..7   (4 C-rows each)
  const int sblk = tid & 31;            // 0..31  (4 S-cols each)

  union RegA { uint4 q; _Float16 h[8]; };
  union RegB4 { unsigned long long q; _Float16 h[4]; };
  RegA rA;
  RegB4 rB[4];                           // rB[r].h[j] = X[c0+4cblk+r][s0+4sblk+j]

  const int T = (C + 31) >> 5;

  auto load_regs = [&](int tile) {
    int c0 = tile * 32;
    bool full = (c0 + 32 <= C);
    // A: 8 contiguous halves of one W row
    if (full) {
      rA.q = *(const uint4*)(Wt + (long long)(o0 + arow) * C + c0 + acol);
    } else {
#pragma unroll
      for (int j = 0; j < 8; ++j) {
        int c = c0 + acol + j;
        rA.h[j] = (c < C) ? Wt[(long long)(o0 + arow) * C + c] : (_Float16)0.0f;
      }
    }
    // B: 4 rows x 4 contiguous halves
#pragma unroll
    for (int r = 0; r < 4; ++r) {
      int cr = c0 + 4 * cblk + r;
      if (full) {
        rB[r].q = *(const unsigned long long*)(X + (long long)cr * S + s0 + 4 * sblk);
      } else {
#pragma unroll
        for (int j = 0; j < 4; ++j)
          rB[r].h[j] = (cr < C) ? X[(long long)cr * S + s0 + 4 * sblk + j] : (_Float16)0.0f;
      }
    }
    // prefetch next-next tile's X rows into L2 (speculative)
    if (c0 + 64 < C)
      __builtin_prefetch((const void*)(X + (long long)(c0 + 32 + 4 * cblk) * S + s0 + 4 * sblk), 0, 1);
  };

  auto store_lds = [&](int buf) {
    *(uint4*)&As[buf][arow][acol] = rA.q;
    // 4x4 in-register transpose -> contiguous K runs in BsT
#pragma unroll
    for (int j = 0; j < 4; ++j) {
      RegB4 col;
#pragma unroll
      for (int r = 0; r < 4; ++r) col.h[r] = rB[r].h[j];
      *(unsigned long long*)&BsT[buf][4 * sblk + j][4 * cblk] = col.q;
    }
  };

  v8f acc[2][2];
#pragma unroll
  for (int u = 0; u < 2; ++u)
#pragma unroll
    for (int v = 0; v < 2; ++v) acc[u][v] = (v8f){};

  load_regs(0);
  store_lds(0);
  __syncthreads();

  for (int t = 0; t < T; ++t) {
    int cur = t & 1;
    if (t + 1 < T) load_regs(t + 1);

    // fragments: contiguous 16-byte LDS reads
    v16h av[2], bv[2];
#pragma unroll
    for (int u = 0; u < 2; ++u) {
      union { uint4 q[2]; v16h h; } fa;
      const _Float16* ap = &As[cur][32 * owv + 16 * u + lm][0];
      fa.q[0] = *(const uint4*)(ap + 8 * gg);
      fa.q[1] = *(const uint4*)(ap + 16 + 8 * gg);
      av[u] = fa.h;
    }
#pragma unroll
    for (int v = 0; v < 2; ++v) {
      union { uint4 q[2]; v16h h; } fb;
      const _Float16* bp = &BsT[cur][32 * swv + 16 * v + lm][0];
      fb.q[0] = *(const uint4*)(bp + 8 * gg);
      fb.q[1] = *(const uint4*)(bp + 16 + 8 * gg);
      bv[v] = fb.h;
    }
#pragma unroll
    for (int u = 0; u < 2; ++u)
#pragma unroll
      for (int v = 0; v < 2; ++v)
        acc[u][v] = __builtin_amdgcn_wmma_f32_16x16x32_f16(false, av[u], false, bv[v],
                                                           (short)0, acc[u][v], false, false);
    __syncthreads();
    if (t + 1 < T) store_lds((t + 1) & 1);
    __syncthreads();
  }

  // D layout: VGPR r -> row r (lanes 0-15) / r+8 (lanes 16-31), col = lane&15.
#pragma unroll
  for (int u = 0; u < 2; ++u)
#pragma unroll
    for (int v = 0; v < 2; ++v)
#pragma unroll
      for (int r = 0; r < 8; ++r) {
        int o = o0 + 32 * owv + 16 * u + r + 8 * gg;
        int s = s0 + 32 * swv + 16 * v + lm;
        Y[(long long)o * S + s] = (_Float16)(acc[u][v][r] + bias[o]);
      }
}

// ---------------------------------------------------------------------------
// BatchNorm (global per-channel stats) + ReLU, in place on f16 activations.
// ---------------------------------------------------------------------------
__global__ __launch_bounds__(WG) void bn_stats_kernel(const _Float16* __restrict__ Y,
                                                      float* __restrict__ stats, int S) {
  __shared__ float r1[WG], r2[WG];
  int o = blockIdx.x;
  const _Float16* yp = Y + (long long)o * S;
  float a = 0.f, b = 0.f;
  for (int s = threadIdx.x; s < S; s += WG) { float v = (float)yp[s]; a += v; b += v * v; }
  r1[threadIdx.x] = a; r2[threadIdx.x] = b;
  __syncthreads();
  for (int st = WG / 2; st > 0; st >>= 1) {
    if (threadIdx.x < st) { r1[threadIdx.x] += r1[threadIdx.x + st];
                            r2[threadIdx.x] += r2[threadIdx.x + st]; }
    __syncthreads();
  }
  if (threadIdx.x == 0) { stats[2 * o] = r1[0]; stats[2 * o + 1] = r2[0]; }
}

__global__ __launch_bounds__(WG) void bn_scale_kernel(const float* __restrict__ stats,
                                                      const float* __restrict__ g,
                                                      const float* __restrict__ be,
                                                      float* __restrict__ scale,
                                                      float* __restrict__ shift,
                                                      int O, float invS) {
  int o = blockIdx.x * WG + threadIdx.x;
  if (o >= O) return;
  float mean = stats[2 * o] * invS;
  float var = stats[2 * o + 1] * invS - mean * mean;
  float sc = g[o] * rsqrtf(var + 1e-5f);
  scale[o] = sc;
  shift[o] = be[o] - mean * sc;
}

__global__ __launch_bounds__(WG) void bn_apply_kernel(_Float16* __restrict__ Y,
                                                      const float* __restrict__ scale,
                                                      const float* __restrict__ shift,
                                                      int S, int total) {
  int t = blockIdx.x * WG + threadIdx.x;
  if (t >= total) return;
  int o = t / S;
  float v = (float)Y[t] * scale[o] + shift[o];
  Y[t] = (_Float16)fmaxf(v, 0.f);
}

// Max over k=32 neighbor axis: (O x BM*32) -> (O x BM)
__global__ __launch_bounds__(WG) void maxpool_kernel(const _Float16* __restrict__ Y,
                                                     _Float16* __restrict__ F,
                                                     int BM, int Stot, int total) {
  int t = blockIdx.x * WG + threadIdx.x;
  if (t >= total) return;
  int o = t / BM, s = t % BM;
  const _Float16* src = Y + (long long)o * Stot + (long long)s * 32;
  float m = (float)src[0];
#pragma unroll
  for (int k = 1; k < 32; ++k) m = fmaxf(m, (float)src[k]);
  F[t] = (_Float16)m;
}

// 3-NN inverse-distance interpolation of coarse features.
__global__ __launch_bounds__(WG) void fp_interp_kernel(const int* __restrict__ idx,
                                                       const float* __restrict__ dist,
                                                       const _Float16* __restrict__ fh,
                                                       _Float16* __restrict__ xo,
                                                       int NlTot, int Nl, int Nh, int NhTot,
                                                       int total) {
  int t = blockIdx.x * WG + threadIdx.x;
  if (t >= total) return;
  int c = t / NlTot, s = t % NlTot;
  int b = s / Nl;
  int base = s * 3;
  float d0 = fmaxf(dist[base + 0], 1e-8f);
  float d1 = fmaxf(dist[base + 1], 1e-8f);
  float d2 = fmaxf(dist[base + 2], 1e-8f);
  float w0 = 1.f / d0, w1 = 1.f / d1, w2 = 1.f / d2;
  float inv = 1.f / (w0 + w1 + w2);
  const _Float16* fr = fh + (long long)c * NhTot + (long long)b * Nh;
  float v = (w0 * (float)fr[idx[base + 0]] +
             w1 * (float)fr[idx[base + 1]] +
             w2 * (float)fr[idx[base + 2]]) * inv;
  xo[t] = (_Float16)v;
}

// Final classifier: out[b,n,o] = sum_c Wc[o,c]*X[c,s] + b2[o]  (13x128, f32 out)
__global__ __launch_bounds__(WG) void head_out_kernel(const _Float16* __restrict__ X,
                                                      const float* __restrict__ Wc,
                                                      const float* __restrict__ b2,
                                                      float* __restrict__ out, int Stot) {
  __shared__ float sW[13 * 128];
  __shared__ float sb[13];
  for (int i = threadIdx.x; i < 13 * 128; i += WG) sW[i] = Wc[i];
  if (threadIdx.x < 13) sb[threadIdx.x] = b2[threadIdx.x];
  __syncthreads();
  int s = blockIdx.x * WG + threadIdx.x;
  if (s >= Stot) return;
  float acc[13];
#pragma unroll
  for (int o = 0; o < 13; ++o) acc[o] = sb[o];
  for (int c = 0; c < 128; ++c) {
    float xv = (float)X[(long long)c * Stot + s];
#pragma unroll
    for (int o = 0; o < 13; ++o) acc[o] += sW[o * 128 + c] * xv;
  }
  float* op = out + (long long)s * 13;
#pragma unroll
  for (int o = 0; o < 13; ++o) op[o] = acc[o];
}

// ---------------------------------------------------------------------------
// Host orchestration
// ---------------------------------------------------------------------------
extern "C" void kernel_launch(void* const* d_in, const int* in_sizes, int n_in,
                              void* d_out, int out_size, void* d_ws, size_t ws_size,
                              hipStream_t stream) {
  (void)in_sizes; (void)out_size; (void)ws_size; (void)n_in;
  constexpr int B = 8, N = 8192, M1 = 2048, M2 = 512;
  constexpr int S1 = B * M1 * 32;   // 524288
  constexpr int S2 = B * M2 * 32;   // 131072
  constexpr int P1 = B * M1;        // 16384
  constexpr int P2 = B * M2;        // 4096
  constexpr int P0 = B * N;         // 65536

  const int OL[11] = {64, 128, 256, 256, 512, 512, 256, 256, 256, 128, 128};
  const int CL[11] = {6, 64, 128, 259, 256, 512, 768, 256, 256, 256, 128};

  const float* pts = (const float*)d_in[0];
  const float *Wf[11], *bf[11], *gf[11], *bef[11];
  int pi = 1;
  for (int l = 0; l < 11; ++l) {
    Wf[l]  = (const float*)d_in[pi++];
    bf[l]  = (const float*)d_in[pi++];
    gf[l]  = (const float*)d_in[pi++];
    bef[l] = (const float*)d_in[pi++];
  }
  const float* Wc = (const float*)d_in[pi++];
  const float* b2 = (const float*)d_in[pi++];

  char* wsp = (char*)d_ws;
  auto alloc = [&](size_t bytes) -> void* {
    void* p = (void*)wsp;
    wsp += (bytes + 255) & ~(size_t)255;
    return p;
  };
  _Float16* Wh[11];
  for (int l = 0; l < 11; ++l) Wh[l] = (_Float16*)alloc((size_t)OL[l] * CL[l] * 2);
  _Float16* feats0 = (_Float16*)alloc((size_t)3 * P0 * 2);
  float* ctr1 = (float*)alloc((size_t)P1 * 3 * 4);
  float* ctr2 = (float*)alloc((size_t)P2 * 3 * 4);
  int* knn1 = (int*)alloc((size_t)P1 * 32 * 4);
  int* knn2 = (int*)alloc((size_t)P2 * 32 * 4);
  int* fpi1 = (int*)alloc((size_t)P1 * 3 * 4);
  float* fpd1 = (float*)alloc((size_t)P1 * 3 * 4);
  int* fpi2 = (int*)alloc((size_t)P0 * 3 * 4);
  float* fpd2 = (float*)alloc((size_t)P0 * 3 * 4);
  _Float16* xcat1 = (_Float16*)alloc((size_t)6 * S1 * 2);
  _Float16* f1 = (_Float16*)alloc((size_t)256 * P1 * 2);
  _Float16* f2 = (_Float16*)alloc((size_t)512 * P2 * 2);
  _Float16* fu1 = (_Float16*)alloc((size_t)256 * P1 * 2);
  float* stats = (float*)alloc((size_t)2 * 512 * 4);
  float* scl = (float*)alloc((size_t)512 * 4);
  float* sft = (float*)alloc((size_t)512 * 4);
  _Float16* bufA = (_Float16*)alloc((size_t)256 * S1 * 2);  // 268 MB arena
  _Float16* bufB = (_Float16*)alloc((size_t)128 * S1 * 2);  // 134 MB arena

  auto cdiv = [](long long a, long long b) { return (unsigned)((a + b - 1) / b); };

  for (int l = 0; l < 11; ++l) {
    int n = OL[l] * CL[l];
    wcvt_kernel<<<cdiv(n, WG), WG, 0, stream>>>(Wf[l], Wh[l], n);
  }
  feats_split_kernel<<<cdiv(3 * P0, WG), WG, 0, stream>>>(pts, feats0, P0, 3 * P0);

  auto run_layer = [&](int l, const _Float16* X, _Float16* Y, int S) {
    dim3 g(S / 128, OL[l] / 64);
    gemm_bias_kernel<<<g, WG, 0, stream>>>(Wh[l], X, bf[l], Y, OL[l], CL[l], S);
    bn_stats_kernel<<<OL[l], WG, 0, stream>>>(Y, stats, S);
    bn_scale_kernel<<<cdiv(OL[l], WG), WG, 0, stream>>>(stats, gf[l], bef[l], scl, sft,
                                                        OL[l], 1.0f / (float)S);
    int total = OL[l] * S;
    bn_apply_kernel<<<cdiv(total, WG), WG, 0, stream>>>(Y, scl, sft, S, total);
  };

  // ---------------- SA1 ----------------
  centers_kernel<<<cdiv(P1, WG), WG, 0, stream>>>(pts, 6, ctr1, M1, N, P1);
  knn_kernel<32><<<cdiv(P1, WG), WG, 0, stream>>>(ctr1, 3, pts, 6, knn1, nullptr, M1, N, P1);
  {
    long long total = (long long)6 * S1;
    sa_concat_kernel<<<cdiv(total, WG), WG, 0, stream>>>(pts, 6, ctr1, feats0, knn1, xcat1,
                                                         M1, N, P0, S1, total);
  }
  run_layer(0, xcat1, bufA, S1);
  run_layer(1, bufA, bufB, S1);
  run_layer(2, bufB, bufA, S1);
  maxpool_kernel<<<cdiv((long long)256 * P1, WG), WG, 0, stream>>>(bufA, f1, P1, S1, 256 * P1);

  // ---------------- SA2 ----------------
  centers_kernel<<<cdiv(P2, WG), WG, 0, stream>>>(ctr1, 3, ctr2, M2, M1, P2);
  knn_kernel<32><<<cdiv(P2, WG), WG, 0, stream>>>(ctr2, 3, ctr1, 3, knn2, nullptr, M2, M1, P2);
  {
    long long total = (long long)259 * S2;
    sa_concat_kernel<<<cdiv(total, WG), WG, 0, stream>>>(ctr1, 3, ctr2, f1, knn2, bufB,
                                                         M2, M1, P1, S2, total);
  }
  run_layer(3, bufB, bufA, S2);
  run_layer(4, bufA, bufB, S2);
  run_layer(5, bufB, bufA, S2);
  maxpool_kernel<<<cdiv((long long)512 * P2, WG), WG, 0, stream>>>(bufA, f2, P2, S2, 512 * P2);

  // ---------------- FP1 (xyz1 <- xyz2) ----------------
  knn_kernel<3><<<cdiv(P1, WG), WG, 0, stream>>>(ctr1, 3, ctr2, 3, fpi1, fpd1, M1, M2, P1);
  fp_interp_kernel<<<cdiv((long long)512 * P1, WG), WG, 0, stream>>>(fpi1, fpd1, f2, bufB,
                                                                     P1, M1, M2, P2, 512 * P1);
  copyh_kernel<<<cdiv((long long)256 * P1, WG), WG, 0, stream>>>(f1, bufB + (size_t)512 * P1,
                                                                 256 * P1);
  run_layer(6, bufB, bufA, P1);
  run_layer(7, bufA, fu1, P1);

  // ---------------- FP2 (xyz <- xyz1) ----------------
  knn_kernel<3><<<cdiv(P0, WG), WG, 0, stream>>>(pts, 6, ctr1, 3, fpi2, fpd2, N, M1, P0);
  fp_interp_kernel<<<cdiv((long long)256 * P0, WG), WG, 0, stream>>>(fpi2, fpd2, fu1, bufA,
                                                                     P0, N, M1, P1, 256 * P0);
  run_layer(8, bufA, bufB, P0);
  run_layer(9, bufB, bufA, P0);   // fu0 = 128 x 65536 in bufA

  // ---------------- Head ----------------
  run_layer(10, bufA, bufB, P0);
  head_out_kernel<<<cdiv(P0, WG), WG, 0, stream>>>(bufB, Wc, b2, (float*)d_out, P0);
}
